// TreeModule_81329500717100
// MI455X (gfx1250) — matrix operand order repaired and checked by reading, not compiled, not run on hardware
//
#include <hip/hip_runtime.h>
#include <math.h>

#define B_   4096
#define D_   2048
#define E_   16
#define C_   1000
#define CP_  1024   // padded C so WMMA tiles need no lane guards

typedef __attribute__((ext_vector_type(16))) __bf16       v16bf;
typedef __attribute__((ext_vector_type(8)))  float        v8f;
typedef __attribute__((ext_vector_type(4)))  unsigned int u32x4;

union BFrag {
    v16bf v;
    u32x4 q[2];
    unsigned short u[16];
};

__device__ __forceinline__ unsigned short f32_to_bf16(float f) {
    unsigned int u = __float_as_uint(f);
    u += 0x7FFFu + ((u >> 16) & 1u);      // round-to-nearest-even
    return (unsigned short)(u >> 16);
}

// ---------------------------------------------------------------------------
// Kernel 0: zero fill (used for out and for expert counters)
// ---------------------------------------------------------------------------
__global__ void zero_kernel(float* __restrict__ p, int n) {
    int i = blockIdx.x * 256 + threadIdx.x;
    if (i < n) p[i] = 0.0f;
}

// ---------------------------------------------------------------------------
// Kernel 1: gating  (choose = x@Wr + br, top-2, softmax, scatter to lists)
// one 64-thread block per sample; thread = (part 0..3, expert 0..15)
// ---------------------------------------------------------------------------
__global__ __launch_bounds__(64) void gate_kernel(
    const float* __restrict__ x,
    const float* __restrict__ Wr,
    const float* __restrict__ br,
    int*   __restrict__ counts,    // [E]
    int*   __restrict__ list,      // [E,B]
    float* __restrict__ wlist)     // [E,B]
{
    const int b    = blockIdx.x;
    const int tid  = threadIdx.x;
    const int e    = tid & 15;
    const int part = tid >> 4;     // 0..3

    const float* xr = x + (size_t)b * D_;
    float s = 0.0f;
    for (int d = part; d < D_; d += 4)
        s = fmaf(xr[d], Wr[d * E_ + e], s);

    __shared__ float red[64];
    red[tid] = s;
    __syncthreads();

    if (tid < 16)
        red[tid] = red[tid] + red[tid + 16] + red[tid + 32] + red[tid + 48] + br[tid];
    __syncthreads();

    if (tid == 0) {
        int   i0 = 0;  float v0 = red[0];
        for (int i = 1; i < E_; ++i) if (red[i] > v0) { v0 = red[i]; i0 = i; }
        int   i1 = -1; float v1 = -INFINITY;
        for (int i = 0; i < E_; ++i) if (i != i0 && red[i] > v1) { v1 = red[i]; i1 = i; }
        // softmax over {v0, v1}, v0 >= v1
        float e0 = 1.0f, e1 = __expf(v1 - v0);
        float inv = 1.0f / (e0 + e1);
        float w0 = e0 * inv, w1 = e1 * inv;

        int p0 = atomicAdd(&counts[i0], 1);
        list[i0 * B_ + p0] = b;  wlist[i0 * B_ + p0] = w0;
        int p1 = atomicAdd(&counts[i1], 1);
        list[i1 * B_ + p1] = b;  wlist[i1 * B_ + p1] = w1;
    }
}

// ---------------------------------------------------------------------------
// Kernel 2: We[e,d,c] fp32  ->  WeT[e,c,d] bf16, C padded to CP_ with zeros
// 32x32 tile transpose through LDS; grid = (E, D/32, CP/32), block = 256
// ---------------------------------------------------------------------------
__global__ __launch_bounds__(256) void convert_We_kernel(
    const float* __restrict__ We,
    unsigned short* __restrict__ WeT)
{
    const int e  = blockIdx.x;
    const int dt = blockIdx.y;
    const int ct = blockIdx.z;

    __shared__ float tile[32][33];

    const int tr = threadIdx.x >> 5;   // 0..7
    const int tc = threadIdx.x & 31;

    #pragma unroll
    for (int i = 0; i < 4; ++i) {
        int d = dt * 32 + tr + i * 8;
        int c = ct * 32 + tc;
        float v = (c < C_) ? We[((size_t)e * D_ + d) * C_ + c] : 0.0f;
        tile[tr + i * 8][tc] = v;
    }
    __syncthreads();

    #pragma unroll
    for (int i = 0; i < 4; ++i) {
        int c = ct * 32 + tr + i * 8;
        int d = dt * 32 + tc;
        WeT[((size_t)e * CP_ + c) * D_ + d] = f32_to_bf16(tile[tc][tr + i * 8]);
    }
}

// ---------------------------------------------------------------------------
// Kernel 2b: x fp32 -> bf16 (row-major copy, [B,D])
// ---------------------------------------------------------------------------
__global__ __launch_bounds__(256) void convert_x_kernel(
    const float* __restrict__ x,
    unsigned short* __restrict__ xb)
{
    int i = blockIdx.x * 256 + threadIdx.x;      // B_*D_/4 threads, 4 elems each
    const float* src = x + (size_t)i * 4;
    unsigned short v0 = f32_to_bf16(src[0]);
    unsigned short v1 = f32_to_bf16(src[1]);
    unsigned short v2 = f32_to_bf16(src[2]);
    unsigned short v3 = f32_to_bf16(src[3]);
    unsigned short* dst = xb + (size_t)i * 4;
    dst[0] = v0; dst[1] = v1; dst[2] = v2; dst[3] = v3;
}

// ---------------------------------------------------------------------------
// Kernel 3: MoE expert GEMM with bf16 WMMA, register double-buffered K loop.
// grid = (B/16, E, CP/512); block = 256 threads = 8 waves.
// Each wave owns FOUR 16x16 subtiles (columns wave*16 + {0,128,256,384});
// all fragments come straight from global (no LDS / barriers in the K loop),
// and the next K-stage's loads are issued before the current stage's WMMAs
// so the scheduler can overlap loads with matrix math (partial loadcnt waits).
// ---------------------------------------------------------------------------
#define LOAD_STAGE(AF, F0, F1, F2, F3, D0)                                   \
    do {                                                                     \
        (AF).q[0] = *(const u32x4*)(aptr + (D0));                            \
        (AF).q[1] = *(const u32x4*)(aptr + (D0) + 8);                        \
        const u32x4* _p;                                                     \
        _p = (const u32x4*)(b0 + (D0)); (F0).q[0] = _p[0]; (F0).q[1] = _p[1];\
        _p = (const u32x4*)(b1 + (D0)); (F1).q[0] = _p[0]; (F1).q[1] = _p[1];\
        _p = (const u32x4*)(b2 + (D0)); (F2).q[0] = _p[0]; (F2).q[1] = _p[1];\
        _p = (const u32x4*)(b3 + (D0)); (F3).q[0] = _p[0]; (F3).q[1] = _p[1];\
    } while (0)

#define WMMA_STAGE(AF, F0, F1, F2, F3)                                       \
    do {                                                                     \
        acc0 = __builtin_amdgcn_wmma_f32_16x16x32_bf16(                      \
                   false, (AF).v, false, (F0).v, (short)0, acc0, false, false);\
        acc1 = __builtin_amdgcn_wmma_f32_16x16x32_bf16(                      \
                   false, (AF).v, false, (F1).v, (short)0, acc1, false, false);\
        acc2 = __builtin_amdgcn_wmma_f32_16x16x32_bf16(                      \
                   false, (AF).v, false, (F2).v, (short)0, acc2, false, false);\
        acc3 = __builtin_amdgcn_wmma_f32_16x16x32_bf16(                      \
                   false, (AF).v, false, (F3).v, (short)0, acc3, false, false);\
    } while (0)

__global__ __launch_bounds__(256) void moe_gemm_kernel(
    const unsigned short* __restrict__ xb,    // [B,D] bf16
    const unsigned short* __restrict__ WeT,   // [E,CP,D] bf16
    const float* __restrict__ be,             // [E,C]
    const int*   __restrict__ counts,         // [E]
    const int*   __restrict__ list,           // [E,B]
    const float* __restrict__ wlist,          // [E,B]
    float* __restrict__ out)                  // [B,C]
{
    const int mt = blockIdx.x;
    const int e  = blockIdx.y;
    const int nt = blockIdx.z;

    const int rows = counts[e];
    if (mt * 16 >= rows) return;              // uniform early-exit

    __shared__ int   rowIdxS[16];
    __shared__ float rowWS[16];

    const int tid = threadIdx.x;
    if (tid < 16) {
        int r = mt * 16 + tid;
        if (r < rows) {
            rowIdxS[tid] = list[e * B_ + r];
            rowWS[tid]   = wlist[e * B_ + r];
        } else {
            rowIdxS[tid] = -1;
            rowWS[tid]   = 0.0f;
        }
    }
    __syncthreads();

    const int wave = tid >> 5;
    const int lane = tid & 31;
    const int half = lane >> 4;               // K-half for A/B, M-half for C/D
    const int l15  = lane & 15;
    const int ncb  = nt * 512 + wave * 16 + l15;   // base padded column

    // A row pointer for this lane (row m = l15 of the gathered tile)
    const int rowA = rowIdxS[l15];
    const unsigned short* aptr =
        xb + (size_t)(rowA < 0 ? 0 : rowA) * D_ + half * 16;

    // B panels: WeT[e][n][d] -> lane n reads 16 contiguous k values
    const unsigned short* b0 = WeT + ((size_t)e * CP_ + ncb +   0) * D_ + half * 16;
    const unsigned short* b1 = WeT + ((size_t)e * CP_ + ncb + 128) * D_ + half * 16;
    const unsigned short* b2 = WeT + ((size_t)e * CP_ + ncb + 256) * D_ + half * 16;
    const unsigned short* b3 = WeT + ((size_t)e * CP_ + ncb + 384) * D_ + half * 16;

    v8f acc0 = {}, acc1 = {}, acc2 = {}, acc3 = {};

    // Two fragment register buffers (ping/pong): A + 4xB each.
    BFrag aX, x0, x1, x2, x3;     // buffer X
    BFrag aY, y0, y1, y2, y3;     // buffer Y

    LOAD_STAGE(aX, x0, x1, x2, x3, 0);

    // NOTE: the final LOAD_STAGE of each iteration reads up to 64B past the
    // logical row; all such reads stay inside d_ws and are never consumed.
    for (int d0 = 0; d0 < D_; d0 += 64) {
        LOAD_STAGE(aY, y0, y1, y2, y3, d0 + 32);
        __builtin_prefetch(b0 + d0 + 128, 0, 1);
        WMMA_STAGE(aX, x0, x1, x2, x3);
        LOAD_STAGE(aX, x0, x1, x2, x3, d0 + 64);
        __builtin_prefetch(b2 + d0 + 128, 0, 1);
        WMMA_STAGE(aY, y0, y1, y2, y3);
    }

    // Epilogue: out[b, n] += w * (acc + be[e, n]); lane guards only AFTER WMMA.
    #pragma unroll
    for (int j = 0; j < 4; ++j) {
        const int ncol = ncb + j * 128;
        if (ncol >= C_) continue;
        const float bev = be[e * C_ + ncol];
        const v8f& acc = (j == 0) ? acc0 : (j == 1) ? acc1 : (j == 2) ? acc2 : acc3;
        #pragma unroll
        for (int r = 0; r < 8; ++r) {
            const int m = r + half * 8;       // C/D layout: VGPR r -> M = r + 8*half
            const int bidx = rowIdxS[m];
            if (bidx < 0) continue;
            atomicAdd(&out[(size_t)bidx * C_ + ncol], rowWS[m] * (acc[r] + bev));
        }
    }
}

// ---------------------------------------------------------------------------
extern "C" void kernel_launch(void* const* d_in, const int* in_sizes, int n_in,
                              void* d_out, int out_size, void* d_ws, size_t ws_size,
                              hipStream_t stream) {
    const float* x  = (const float*)d_in[0];   // [B,D]
    const float* Wr = (const float*)d_in[1];   // [D,E]
    const float* br = (const float*)d_in[2];   // [E]
    const float* We = (const float*)d_in[3];   // [E,D,C]
    const float* be = (const float*)d_in[4];   // [E,C]
    float* out = (float*)d_out;                // [B,1,C] flat

    // workspace layout
    const size_t WET_BYTES = (size_t)E_ * CP_ * D_ * 2;       // 67,108,864
    const size_t XB_OFF    = WET_BYTES;
    const size_t XB_BYTES  = (size_t)B_ * D_ * 2;             // 16,777,216
    const size_t CNT_OFF   = XB_OFF + XB_BYTES;
    const size_t LIST_OFF  = CNT_OFF + 256;
    const size_t WLIST_OFF = LIST_OFF + (size_t)E_ * B_ * 4;
    const size_t WS_NEEDED = WLIST_OFF + (size_t)E_ * B_ * 4;
    if (ws_size < WS_NEEDED) return;

    char* ws = (char*)d_ws;
    unsigned short* WeT = (unsigned short*)ws;
    unsigned short* xb  = (unsigned short*)(ws + XB_OFF);
    int*   counts = (int*)  (ws + CNT_OFF);
    int*   list   = (int*)  (ws + LIST_OFF);
    float* wlist  = (float*)(ws + WLIST_OFF);

    // 0) zero output accumulator and expert counters
    zero_kernel<<<(B_ * C_ + 255) / 256, 256, 0, stream>>>(out, B_ * C_);
    zero_kernel<<<1, 256, 0, stream>>>((float*)counts, E_);

    // 1) gating + dispatch
    gate_kernel<<<B_, 64, 0, stream>>>(x, Wr, br, counts, list, wlist);

    // 2) weight convert/transpose to bf16 [E, CP, D]; x -> bf16 [B, D]
    convert_We_kernel<<<dim3(E_, D_ / 32, CP_ / 32), 256, 0, stream>>>(We, WeT);
    convert_x_kernel<<<(B_ * D_ / 4 + 255) / 256, 256, 0, stream>>>(x, xb);

    // 3) selected-expert GEMM + weighted combine
    moe_gemm_kernel<<<dim3(B_ / 16, E_, CP_ / 512), 256, 0, stream>>>(
        xb, WeT, be, counts, list, wlist, out);
}
